// GPTLanguageModel_21414706937899
// MI455X (gfx1250) — matrix operand order repaired and checked by reading, not compiled, not run on hardware
//
#include <hip/hip_runtime.h>

// ---------------- model constants ----------------
#define TT   2048
#define CC   512
#define NH   8
#define HS   64
#define NB   2
#define MR   (NB*TT)     // 4096 rows
#define NL   4
#define FF   (4*CC)      // 2048
#define VOC  32000

typedef __attribute__((ext_vector_type(16))) _Float16 v16h;
typedef __attribute__((ext_vector_type(8)))  _Float16 v8h;
typedef __attribute__((ext_vector_type(8)))  float    v8f;

static __device__ __forceinline__ v8f wmma16(v16h a, v16h b, v8f c) {
  // D = A(16x32 f16) * B(32x16 f16) + C(16x16 f32)
  return __builtin_amdgcn_wmma_f32_16x16x32_f16(false, a, false, b, (short)0, c, false, false);
}

static __device__ __forceinline__ v16h load_afrag(const _Float16* rowp, int kbA) {
  v8h lo = *(const v8h*)(rowp + kbA);
  v8h hh = *(const v8h*)(rowp + kbA + 8);
  v16h a;
#pragma unroll
  for (int i = 0; i < 8; i++) { a[i] = lo[i]; a[8 + i] = hh[i]; }
  return a;
}

// ---------------- embedding ----------------
__global__ void embed_kernel(const int* __restrict__ toks,
                             const float* __restrict__ tok_emb,
                             const float* __restrict__ pos_emb,
                             float* __restrict__ x) {
  size_t idx = (size_t)blockIdx.x * 256 + threadIdx.x;   // MR*CC total
  int c = idx & (CC - 1);
  size_t bt = idx >> 9;
  int t = bt & (TT - 1);
  int tok = toks[bt];
  x[idx] = tok_emb[(size_t)tok * CC + c] + pos_emb[(size_t)t * CC + c];
}

// ---------------- weight pack: dst[n*dld + k] = f16(src[k*sld + n]) ----------------
__global__ void pack_bt(const float* __restrict__ src, _Float16* __restrict__ dst,
                        int K, int N, int sld, int dld) {
  int idx = blockIdx.x * 256 + threadIdx.x;
  if (idx >= K * N) return;
  int k = idx % K, n = idx / K;
  dst[(size_t)n * dld + k] = (_Float16)src[(size_t)k * sld + n];
}

// ---------------- layernorm (row of 512), f16 output ----------------
__global__ __launch_bounds__(256) void ln_kernel(const float* __restrict__ x,
                                                 const float* __restrict__ g,
                                                 const float* __restrict__ b,
                                                 _Float16* __restrict__ out) {
  int row = blockIdx.x, tid = threadIdx.x;
  const float* xr = x + (size_t)row * CC;
  float v0 = xr[tid], v1 = xr[tid + 256];
  __shared__ float s1[256], s2[256];
  s1[tid] = v0 + v1;
  s2[tid] = v0 * v0 + v1 * v1;
  __syncthreads();
  for (int st = 128; st > 0; st >>= 1) {
    if (tid < st) { s1[tid] += s1[tid + st]; s2[tid] += s2[tid + st]; }
    __syncthreads();
  }
  float mu  = s1[0] * (1.0f / CC);
  float var = s2[0] * (1.0f / CC) - mu * mu;
  float rs  = rsqrtf(var + 1e-5f);
  out[(size_t)row * CC + tid]       = (_Float16)((v0 - mu) * rs * g[tid]       + b[tid]);
  out[(size_t)row * CC + tid + 256] = (_Float16)((v1 - mu) * rs * g[tid + 256] + b[tid + 256]);
}

// ---------------- attention mask bits: (local|global|random)&causal ----------------
__global__ void mask_kernel(const int* __restrict__ rc, unsigned* __restrict__ mb) {
  int idx = blockIdx.x * 256 + threadIdx.x;     // NH*TT*64 words
  int w = idx & 63;
  int t = (idx >> 6) & (TT - 1);
  int h = idx >> 17;
  unsigned word = 0;
  int s0 = w * 32;
  for (int j = 0; j < 32; j++) {
    int s = s0 + j;
    bool loc = (s >= t - 63) && (s <= t);
    bool gl  = (t < 16) || (s < 16);
    bool cz  = (s <= t);
    if ((loc || gl) && cz) word |= 1u << j;
  }
  const int* r = rc + ((size_t)h * TT + t) * 16;
  for (int i = 0; i < 16; i++) {
    int s = r[i];
    if (s <= t && (s >> 5) == w) word |= 1u << (s & 31);
  }
  mb[idx] = word;
}

// ---------------- qkv reshape: flat [bt, h*64+d] -> q/k [bh,t,d], vT [bh,d,t] ----------------
__global__ void qkv_reshape(const _Float16* __restrict__ qf, const _Float16* __restrict__ kf,
                            const _Float16* __restrict__ vf,
                            _Float16* __restrict__ q, _Float16* __restrict__ k,
                            _Float16* __restrict__ vt) {
  size_t idx = (size_t)blockIdx.x * 256 + threadIdx.x;   // MR*CC
  int c = idx & (CC - 1);
  size_t bt = idx >> 9;
  int t = bt & (TT - 1);
  int b = bt >> 11;
  int h = c >> 6, d = c & 63;
  int bh = b * NH + h;
  size_t qi = ((size_t)bh * TT + t) * HS + d;
  q[qi] = qf[idx];
  k[qi] = kf[idx];
  vt[((size_t)bh * HS + d) * TT + t] = vf[idx];
}

// ---------------- generic WMMA GEMM: C[M,N] = A[M,K](f16) * BT[N,K](f16) ----------------
// workgroup 256 thr = 8 waves; block tile 64(M) x 256(N);
// each wave: 32(M) x 64(N) => 2 A-frags x 4 B-frags = 8 WMMA per K-step
// (each B fragment feeds two WMMAs; halves B traffic per FLOP vs 16-row tiles)
__global__ __launch_bounds__(256) void gemm_wmma(
    const _Float16* __restrict__ A, const _Float16* __restrict__ BT,
    const float* __restrict__ bias, const float* __restrict__ res,
    float* __restrict__ outF, _Float16* __restrict__ outH,
    int M, int N, int K, int relu)
{
  int tid = threadIdx.x;
  int lane = tid & 31, wave = tid >> 5;
  int lm = lane & 15, hi = lane >> 4;
  int wm = wave & 1, wn = wave >> 1;
  int row0 = blockIdx.y * 64 + wm * 32;
  int col0 = blockIdx.x * 256 + wn * 64;
  int kbA = hi ? 16 : 0;    // A frag: lanes0-15 K=0..15 ; lanes16-31 K=16..31
  int kbB = hi ? 16 : 0;    // B frag: lanes0-15 K=0..15 ; lanes16-31 K=16..31
  const _Float16* Ar0 = A + (size_t)(row0 + lm) * K;
  const _Float16* Ar1 = A + (size_t)(row0 + 16 + lm) * K;

  v8f acc[2][4];
#pragma unroll
  for (int m = 0; m < 2; m++)
#pragma unroll
    for (int i = 0; i < 4; i++) acc[m][i] = (v8f)0.0f;

#pragma unroll 2
  for (int k0 = 0; k0 < K; k0 += 32) {
    v16h a0 = load_afrag(Ar0 + k0, kbA);
    v16h a1 = load_afrag(Ar1 + k0, kbA);
#pragma unroll
    for (int nt = 0; nt < 4; nt++) {
      const _Float16* Bp = BT + (size_t)(col0 + nt * 16 + lm) * K + k0 + kbB;
      v16h b = *(const v16h*)Bp;            // 16 contiguous K values
      acc[0][nt] = wmma16(a0, b, acc[0][nt]);
      acc[1][nt] = wmma16(a1, b, acc[1][nt]);
    }
  }
#pragma unroll
  for (int m = 0; m < 2; m++) {
#pragma unroll
    for (int nt = 0; nt < 4; nt++) {
#pragma unroll
      for (int r = 0; r < 8; r++) {
        int row = row0 + m * 16 + r + 8 * hi;
        int col = col0 + nt * 16 + lm;
        float v = acc[m][nt][r];
        if (bias) v += bias[col];
        if (relu) v = fmaxf(v, 0.0f);
        size_t o = (size_t)row * N + col;
        if (res)  v += res[o];
        if (outF) outF[o] = v;
        if (outH) outH[o] = (_Float16)v;
      }
    }
  }
}

// ---------------- flash attention: 1 wave per (b,h, 16 query rows) ----------------
__global__ __launch_bounds__(256) void attn_kernel(
    const _Float16* __restrict__ q, const _Float16* __restrict__ k,
    const _Float16* __restrict__ vt, const unsigned* __restrict__ mb,
    _Float16* __restrict__ out)
{
  __shared__ __align__(32) _Float16 pl_all[8][16 * 32];
  int tid = threadIdx.x;
  int wave = tid >> 5, lane = tid & 31;
  int lm = lane & 15, hi = lane >> 4;
  int W = blockIdx.x * 8 + wave;          // 0..2047
  int tile = W & 127;
  int bh = W >> 7;
  int h = bh & (NH - 1), b = bh >> 3;
  int t0 = tile * 16;
  const _Float16* qp = q  + (size_t)bh * TT * HS;
  const _Float16* kp = k  + (size_t)bh * TT * HS;
  const _Float16* vp = vt + (size_t)bh * HS * TT;
  const unsigned* mbp = mb + (size_t)h * TT * 64;
  _Float16* pl = pl_all[wave];
  int kbA = hi ? 16 : 0, kbB = hi ? 16 : 0;

  // Q A-fragments for the two K=32 chunks of D=64
  const _Float16* qr = qp + (size_t)(t0 + lm) * HS;
  v16h aq0 = load_afrag(qr, kbA);
  v16h aq1 = load_afrag(qr + 32, kbA);

  v8f acc[4];
#pragma unroll
  for (int i = 0; i < 4; i++) acc[i] = (v8f)0.0f;
  float m_i[8], l_i[8];
#pragma unroll
  for (int r = 0; r < 8; r++) { m_i[r] = -1e30f; l_i[r] = 0.0f; }

  const float scale = 0.125f;   // HS^-0.5
  int tHi = t0 + 15;

  for (int s0 = 0; s0 <= tHi; s0 += 32) {
    int w0 = s0 >> 5;
    // --- scores: two 16x16 tiles over this 32-key chunk ---
    v8f c0 = (v8f)0.0f, c1 = (v8f)0.0f;
    {
      const _Float16* kr = kp + (size_t)(s0 + lm) * HS;
      v16h b0 = *(const v16h*)(kr + kbB);
      v16h b1 = *(const v16h*)(kr + 32 + kbB);
      c0 = wmma16(aq0, b0, c0);
      c0 = wmma16(aq1, b1, c0);
    }
    {
      const _Float16* kr = kp + (size_t)(s0 + 16 + lm) * HS;
      v16h b0 = *(const v16h*)(kr + kbB);
      v16h b1 = *(const v16h*)(kr + 32 + kbB);
      c1 = wmma16(aq0, b0, c1);
      c1 = wmma16(aq1, b1, c1);
    }
    // --- mask + online softmax stats (rows striped across 16-lane halves) ---
#pragma unroll
    for (int r = 0; r < 8; r++) {
      int t = t0 + r + 8 * hi;
      unsigned mw = mbp[(size_t)t * 64 + w0];
      int sA = s0 + lm, sB = s0 + 16 + lm;
      float v0 = ((mw >> (sA & 31)) & 1u) ? c0[r] * scale : -1e30f;
      float v1 = ((mw >> (sB & 31)) & 1u) ? c1[r] * scale : -1e30f;
      float rm = fmaxf(v0, v1);
      for (int d = 1; d < 16; d <<= 1) rm = fmaxf(rm, __shfl_xor(rm, d, 32));
      float mn = fmaxf(m_i[r], rm);
      float al = __expf(m_i[r] - mn);
      m_i[r] = mn;
      float p0 = __expf(v0 - mn), p1 = __expf(v1 - mn);
      float ps = p0 + p1;
      for (int d = 1; d < 16; d <<= 1) ps += __shfl_xor(ps, d, 32);
      l_i[r] = l_i[r] * al + ps;
      acc[0][r] *= al; acc[1][r] *= al; acc[2][r] *= al; acc[3][r] *= al;
      int rowm = r + 8 * hi;
      pl[rowm * 32 + lm]      = (_Float16)p0;
      pl[rowm * 32 + 16 + lm] = (_Float16)p1;
    }
    asm volatile("s_wait_dscnt 0" ::: "memory");
    // --- P as A-fragment from LDS, then O += P(16x32) @ V(32x64) ---
    v16h pa = load_afrag(pl + lm * 32, kbA);
#pragma unroll
    for (int dt = 0; dt < 4; dt++) {
      const _Float16* vr = vp + (size_t)(dt * 16 + lm) * TT + s0 + kbB;
      v16h bv = *(const v16h*)vr;           // contiguous along keys (V^T)
      acc[dt] = wmma16(pa, bv, acc[dt]);
    }
  }

  // --- normalize and store (concat heads): out[b,t, h*64+d] ---
#pragma unroll
  for (int r = 0; r < 8; r++) {
    float inv = 1.0f / l_i[r];
    int t = t0 + r + 8 * hi;
#pragma unroll
    for (int dt = 0; dt < 4; dt++) {
      int col = h * HS + dt * 16 + lm;
      out[((size_t)(b * TT + t)) * CC + col] = (_Float16)(acc[dt][r] * inv);
    }
  }
}

// ---------------- host orchestration ----------------
extern "C" void kernel_launch(void* const* d_in, const int* in_sizes, int n_in,
                              void* d_out, int out_size, void* d_ws, size_t ws_size,
                              hipStream_t stream) {
  const float* tok_emb = (const float*)d_in[0];
  const float* pos_emb = (const float*)d_in[1];
  const float* Wq    = (const float*)d_in[2];
  const float* Wk    = (const float*)d_in[3];
  const float* Wv    = (const float*)d_in[4];
  const float* Wproj = (const float*)d_in[5];
  const float* bproj = (const float*)d_in[6];
  const float* ln1_g = (const float*)d_in[7];
  const float* ln1_b = (const float*)d_in[8];
  const float* ln2_g = (const float*)d_in[9];
  const float* ln2_b = (const float*)d_in[10];
  const float* W1    = (const float*)d_in[11];
  const float* b1    = (const float*)d_in[12];
  const float* W2    = (const float*)d_in[13];
  const float* b2    = (const float*)d_in[14];
  const float* lnf_g = (const float*)d_in[15];
  const float* lnf_b = (const float*)d_in[16];
  const float* Wout  = (const float*)d_in[17];
  const float* bout  = (const float*)d_in[18];
  const int*   toks  = (const int*)d_in[19];
  const int*   rcols = (const int*)d_in[20];

  char* ws = (char*)d_ws;
  size_t off = 0;
  auto carve = [&](size_t bytes) -> char* {
    char* p = ws + off;
    off = (off + bytes + 255) & ~(size_t)255;
    return p;
  };
  const size_t H2 = sizeof(_Float16);
  _Float16* wqT  = (_Float16*)carve((size_t)NL * CC * CC * H2);
  _Float16* wkT  = (_Float16*)carve((size_t)NL * CC * CC * H2);
  _Float16* wvT  = (_Float16*)carve((size_t)NL * CC * CC * H2);
  _Float16* wpT  = (_Float16*)carve((size_t)NL * CC * CC * H2);
  _Float16* w1T  = (_Float16*)carve((size_t)NL * CC * FF * H2);
  _Float16* w2T  = (_Float16*)carve((size_t)NL * CC * FF * H2);
  _Float16* woT  = (_Float16*)carve((size_t)VOC * CC * H2);
  float*    x    = (float*)   carve((size_t)MR * CC * 4);
  _Float16* xn   = (_Float16*)carve((size_t)MR * CC * H2);
  _Float16* qf   = (_Float16*)carve((size_t)MR * CC * H2);
  _Float16* kf   = (_Float16*)carve((size_t)MR * CC * H2);
  _Float16* vf   = (_Float16*)carve((size_t)MR * CC * H2);
  _Float16* qh   = (_Float16*)carve(((size_t)MR * CC + 8192) * H2);
  _Float16* kh   = (_Float16*)carve(((size_t)MR * CC + 8192) * H2);
  _Float16* vth  = (_Float16*)carve(((size_t)MR * CC + 8192) * H2);
  _Float16* attnb= (_Float16*)carve((size_t)MR * CC * H2);
  _Float16* h16  = (_Float16*)carve((size_t)MR * FF * H2);
  unsigned* mbits= (unsigned*)carve((size_t)NH * TT * 64 * 4);

  // zero tail pads (attention key-chunk overreads land here; must be 0, not NaN)
  hipMemsetAsync(kh  + (size_t)MR * CC, 0, 8192 * H2, stream);
  hipMemsetAsync(vth + (size_t)MR * CC, 0, 8192 * H2, stream);

  // ---- weight packing (f32 -> f16, transposed to BT[n][k]) ----
  auto packT = [&](const float* src, _Float16* dst, int K, int N, int sld, int dld) {
    pack_bt<<<(K * N + 255) / 256, 256, 0, stream>>>(src, dst, K, N, sld, dld);
  };
  for (int l = 0; l < NL; l++) {
    for (int h = 0; h < NH; h++) {
      size_t so = ((size_t)(l * NH + h) * CC) * HS;
      size_t dofs = (size_t)l * CC * CC + (size_t)h * HS * CC;
      packT(Wq + so, wqT + dofs, CC, HS, HS, CC);
      packT(Wk + so, wkT + dofs, CC, HS, HS, CC);
      packT(Wv + so, wvT + dofs, CC, HS, HS, CC);
    }
    packT(Wproj + (size_t)l * CC * CC, wpT + (size_t)l * CC * CC, CC, CC, CC, CC);
    packT(W1 + (size_t)l * CC * FF, w1T + (size_t)l * CC * FF, CC, FF, FF, CC);
    packT(W2 + (size_t)l * FF * CC, w2T + (size_t)l * FF * CC, FF, CC, CC, FF);
  }
  packT(Wout, woT, CC, VOC, VOC, CC);

  // ---- embedding ----
  embed_kernel<<<(MR * CC) / 256, 256, 0, stream>>>(toks, tok_emb, pos_emb, x);

  // ---- transformer layers ----
  for (int l = 0; l < NL; l++) {
    ln_kernel<<<MR, 256, 0, stream>>>(x, ln1_g + l * CC, ln1_b + l * CC, xn);
    gemm_wmma<<<dim3(CC / 256, MR / 64), 256, 0, stream>>>(
        xn, wqT + (size_t)l * CC * CC, nullptr, nullptr, nullptr, qf, MR, CC, CC, 0);
    gemm_wmma<<<dim3(CC / 256, MR / 64), 256, 0, stream>>>(
        xn, wkT + (size_t)l * CC * CC, nullptr, nullptr, nullptr, kf, MR, CC, CC, 0);
    gemm_wmma<<<dim3(CC / 256, MR / 64), 256, 0, stream>>>(
        xn, wvT + (size_t)l * CC * CC, nullptr, nullptr, nullptr, vf, MR, CC, CC, 0);
    qkv_reshape<<<(MR * CC) / 256, 256, 0, stream>>>(qf, kf, vf, qh, kh, vth);
    mask_kernel<<<(NH * TT * 64) / 256, 256, 0, stream>>>(
        rcols + (size_t)l * NH * TT * 16, mbits);
    attn_kernel<<<256, 256, 0, stream>>>(qh, kh, vth, mbits, attnb);
    gemm_wmma<<<dim3(CC / 256, MR / 64), 256, 0, stream>>>(
        attnb, wpT + (size_t)l * CC * CC, bproj + l * CC, x, x, nullptr, MR, CC, CC, 0);
    ln_kernel<<<MR, 256, 0, stream>>>(x, ln2_g + l * CC, ln2_b + l * CC, xn);
    gemm_wmma<<<dim3(FF / 256, MR / 64), 256, 0, stream>>>(
        xn, w1T + (size_t)l * CC * FF, b1 + l * FF, nullptr, nullptr, h16, MR, FF, CC, 1);
    gemm_wmma<<<dim3(CC / 256, MR / 64), 256, 0, stream>>>(
        h16, w2T + (size_t)l * CC * FF, b2 + l * CC, x, x, nullptr, MR, CC, FF, 0);
  }

  // ---- final LN + logits ----
  ln_kernel<<<MR, 256, 0, stream>>>(x, lnf_g, lnf_b, xn);
  gemm_wmma<<<dim3(VOC / 256, MR / 64), 256, 0, stream>>>(
      xn, woT, bout, nullptr, (float*)d_out, nullptr, MR, VOC, CC, 0);
}
